// DecoderAttention_19593640804530
// MI455X (gfx1250) — matrix operand order repaired and checked by reading, not compiled, not run on hardware
//
#include <hip/hip_runtime.h>
#include <utility>

// ---------------------------------------------------------------------------
// Problem constants (from the reference)
// ---------------------------------------------------------------------------
#define BB     64
#define HW     196
#define FEAT   2048
#define EMB    512
#define HID    1024
#define ATT    512
#define VOCAB  10000
#define TT     20
#define STEPS  19

typedef __attribute__((ext_vector_type(16))) __bf16        v16bf;
typedef __attribute__((ext_vector_type(8)))  float         v8f;
typedef __attribute__((ext_vector_type(4)))  unsigned int  v4u;

// float -> bf16 round-to-nearest-even
__device__ __forceinline__ unsigned short f2bf(float x) {
    unsigned u = __float_as_uint(x);
    u += 0x7FFFu + ((u >> 16) & 1u);
    return (unsigned short)(u >> 16);
}
__device__ __forceinline__ unsigned pack2bf(float lo, float hi) {
    return (unsigned)f2bf(lo) | ((unsigned)f2bf(hi) << 16);
}

// ---------------------------------------------------------------------------
// WMMA GEMM:  C[M,N] = A[M,K] * W[N,K]^T  (+ bias, or += C if beta)
//
// Requirements honored by every call site below:
//   M % 64 == 0, K % 32 == 0, lda/ldb % 4 == 0, A/W 16B-aligned.
//   Only N may be ragged (VOCAB=10000): handled branch-free by index-clamp
//   + 0/1 mask multiply (no divergent exec save/restore around loads).
//
// Block tile 64(M) x 128(N), K step 32. 8 waves in 4(M) x 2(N); each wave
// computes four 16x16 tiles -> one A fragment feeds 4 WMMAs per K step.
// ---------------------------------------------------------------------------
#define TM 64
#define TN 128
#define TK 32
#define LDSP 40   // ushorts per LDS row (80 B: 16B-aligned frags, bank-spread)

__global__ __launch_bounds__(256)
void gemm_wmma_bf16(const float* __restrict__ A, int lda,
                    const float* __restrict__ W, int ldb,
                    const float* __restrict__ bias,
                    float* __restrict__ C, int ldc,
                    int M, int N, int K, int beta)
{
    (void)M;
    __shared__ alignas(16) unsigned short sA[TM * LDSP];
    __shared__ alignas(16) unsigned short sB[TN * LDSP];

    const int tid  = threadIdx.x;
    const int wid  = tid >> 5;
    const int lane = tid & 31;
    const int wm   = wid & 3;            // wave row (M): rows wm*16..
    const int wn   = wid >> 2;           // wave col (N): cols wn*64..
    const int row0 = blockIdx.y * TM;
    const int col0 = blockIdx.x * TN;
    const int hlf  = lane >> 4;
    const int l16  = lane & 15;

    // ---- staging assignment (fixed across K loop) -----------------------
    // A tile: 64 rows x 8 float4; thread handles rows (tid>>3) and +32.
    // W tile: 128 rows x 8 float4; thread handles rows (tid>>3) + 32*t.
    const int c4  = (tid & 7) * 4;       // float column within K slab
    const int rr  = tid >> 3;            // 0..31
    const float* Ap0 = A + (size_t)(row0 + rr) * lda + c4;
    const float* Ap1 = Ap0 + (size_t)32 * lda;
    unsigned short* sAp0 = sA + rr * LDSP + c4;
    unsigned short* sAp1 = sAp0 + 32 * LDSP;

    const float* Wp[4];
    float        wmsk[4];
    unsigned short* sBp[4];
    #pragma unroll
    for (int t = 0; t < 4; ++t) {
        int r  = col0 + rr + 32 * t;
        int rc = (r < N) ? r : (N - 1);            // clamped, always valid
        Wp[t]   = W + (size_t)rc * ldb + c4;
        wmsk[t] = (r < N) ? 1.f : 0.f;
        sBp[t]  = sB + (rr + 32 * t) * LDSP + c4;
    }

    // ---- accumulators: 4 N-tiles per wave -------------------------------
    // C/D layout: lane owns column n (= l16 of tile), rows r + 8*hlf.
    int ncol[4];
    #pragma unroll
    for (int t = 0; t < 4; ++t) ncol[t] = col0 + wn * 64 + t * 16 + l16;
    const int mrow = row0 + wm * 16 + 8 * hlf;     // + r (0..7)

    v8f acc[4];
    if (beta) {
        #pragma unroll
        for (int t = 0; t < 4; ++t) {
            int nc = (ncol[t] < N) ? ncol[t] : (N - 1);
            float msk = (ncol[t] < N) ? 1.f : 0.f;
            #pragma unroll
            for (int r = 0; r < 8; ++r)
                acc[t][r] = C[(size_t)(mrow + r) * ldc + nc] * msk;
        }
    } else {
        #pragma unroll
        for (int t = 0; t < 4; ++t) {
            float b0 = 0.f;
            if (bias) {
                int nc = (ncol[t] < N) ? ncol[t] : (N - 1);
                b0 = bias[nc] * ((ncol[t] < N) ? 1.f : 0.f);
            }
            #pragma unroll
            for (int r = 0; r < 8; ++r) acc[t][r] = b0;
        }
    }

    // ---- main K loop -----------------------------------------------------
    for (int k0 = 0; k0 < K; k0 += TK) {
        // global loads for this slab (b128, unconditional)
        float4 a0 = *(const float4*)(Ap0 + k0);
        float4 a1 = *(const float4*)(Ap1 + k0);
        float4 w0 = *(const float4*)(Wp[0] + k0);
        float4 w1 = *(const float4*)(Wp[1] + k0);
        float4 w2 = *(const float4*)(Wp[2] + k0);
        float4 w3 = *(const float4*)(Wp[3] + k0);
        // prefetch next slab into L2 (speculative: OOB silently dropped)
        __builtin_prefetch(Ap0 + k0 + TK, 0, 1);
        __builtin_prefetch(Wp[0] + k0 + TK, 0, 1);

        __syncthreads();   // previous iteration's fragment reads complete

        { uint2 p; p.x = pack2bf(a0.x, a0.y); p.y = pack2bf(a0.z, a0.w);
          *(uint2*)sAp0 = p; }
        { uint2 p; p.x = pack2bf(a1.x, a1.y); p.y = pack2bf(a1.z, a1.w);
          *(uint2*)sAp1 = p; }
        { uint2 p; p.x = pack2bf(w0.x * wmsk[0], w0.y * wmsk[0]);
          p.y = pack2bf(w0.z * wmsk[0], w0.w * wmsk[0]); *(uint2*)sBp[0] = p; }
        { uint2 p; p.x = pack2bf(w1.x * wmsk[1], w1.y * wmsk[1]);
          p.y = pack2bf(w1.z * wmsk[1], w1.w * wmsk[1]); *(uint2*)sBp[1] = p; }
        { uint2 p; p.x = pack2bf(w2.x * wmsk[2], w2.y * wmsk[2]);
          p.y = pack2bf(w2.z * wmsk[2], w2.w * wmsk[2]); *(uint2*)sBp[2] = p; }
        { uint2 p; p.x = pack2bf(w3.x * wmsk[3], w3.y * wmsk[3]);
          p.y = pack2bf(w3.z * wmsk[3], w3.w * wmsk[3]); *(uint2*)sBp[3] = p; }

        __syncthreads();   // staging visible to all waves

        // A fragment (ISA 16-bit A 16x32 layout): lane row = l16,
        // 16B chunks at K = 8*hlf and K = 16 + 8*hlf.
        const unsigned short* pa = sA + (wm * 16 + l16) * LDSP;
        union { v16bf v; v4u q[2]; } af;
        af.q[0] = *(const v4u*)(pa + 8 * hlf);
        af.q[1] = *(const v4u*)(pa + 16 + 8 * hlf);

        // B fragments: lane holds W row l16 of each tile, K = 8*hlf + 0..7, 16 + 8*hlf + 0..7
        #pragma unroll
        for (int t = 0; t < 4; ++t) {
            const unsigned short* pb = sB + (wn * 64 + t * 16 + l16) * LDSP
                                          + 8 * hlf;
            union { v16bf v; v4u q[2]; } bf;
            bf.q[0] = *(const v4u*)(pb);
            bf.q[1] = *(const v4u*)(pb + 16);
            acc[t] = __builtin_amdgcn_wmma_f32_16x16x32_bf16(
                         false, af.v, false, bf.v, (short)0, acc[t],
                         false, false);
        }
    }

    // ---- epilogue --------------------------------------------------------
    #pragma unroll
    for (int t = 0; t < 4; ++t) {
        if (ncol[t] < N) {
            #pragma unroll
            for (int r = 0; r < 8; ++r)
                C[(size_t)(mrow + r) * ldc + ncol[t]] = acc[t][r];
        }
    }
}

// ---------------------------------------------------------------------------
// emb[s][b][:] = E[captions[b][s]][:]
// ---------------------------------------------------------------------------
__global__ void embed_kernel(const float* __restrict__ E,
                             const int* __restrict__ captions,
                             float* __restrict__ emb)
{
    int s = blockIdx.x, b = blockIdx.y;
    int tok = captions[b * TT + s];
    const float* src = E + (size_t)tok * EMB;
    float* dst = emb + ((size_t)s * BB + b) * EMB;
    for (int e = threadIdx.x; e < EMB; e += blockDim.x) dst[e] = src[e];
}

// ---------------------------------------------------------------------------
// fmean[b][e] = mean over HW of f[b*HW+hw][e]
// ---------------------------------------------------------------------------
__global__ void mean_kernel(const float* __restrict__ f, float* __restrict__ fm)
{
    int b = blockIdx.x;
    for (int e = threadIdx.x; e < EMB; e += blockDim.x) {
        float s = 0.f;
        for (int hw = 0; hw < HW; ++hw)
            s += f[((size_t)b * HW + hw) * EMB + e];
        fm[(size_t)b * EMB + e] = s * (1.0f / (float)HW);
    }
}

// ---------------------------------------------------------------------------
// Bahdanau attention: one block per batch element.
// ---------------------------------------------------------------------------
__global__ __launch_bounds__(256)
void attn_kernel(const float* __restrict__ fa, const float* __restrict__ ha,
                 const float* __restrict__ f,  const float* __restrict__ Vv,
                 const float* __restrict__ bV, float* __restrict__ ctx)
{
    __shared__ float s_ha[ATT];
    __shared__ float s_v[ATT];
    __shared__ float s_w[HW];
    __shared__ float s_red[256];

    const int b = blockIdx.x, tid = threadIdx.x;
    const int wid = tid >> 5, lane = tid & 31;

    for (int i = tid; i < ATT; i += 256) {
        s_ha[i] = ha[(size_t)b * ATT + i];
        s_v[i]  = Vv[i];
    }
    __syncthreads();

    // one wave per HW row: lane-parallel dot + shuffle reduce
    for (int hw = wid; hw < HW; hw += 8) {
        const float* row = fa + ((size_t)b * HW + hw) * ATT;
        float acc = 0.f;
        for (int a = lane; a < ATT; a += 32)
            acc += s_v[a] * tanhf(row[a] + s_ha[a]);
        for (int off = 16; off > 0; off >>= 1)
            acc += __shfl_xor(acc, off, 32);
        if (lane == 0) s_w[hw] = acc + bV[0];
    }
    __syncthreads();

    // softmax over HW (block tree reduce)
    float lv = (tid < HW) ? s_w[tid] : -3.4e38f;
    s_red[tid] = lv;  __syncthreads();
    for (int off = 128; off > 0; off >>= 1) {
        if (tid < off) s_red[tid] = fmaxf(s_red[tid], s_red[tid + off]);
        __syncthreads();
    }
    float mx = s_red[0];  __syncthreads();
    float ex = (tid < HW) ? __expf(lv - mx) : 0.f;
    s_red[tid] = ex;  __syncthreads();
    for (int off = 128; off > 0; off >>= 1) {
        if (tid < off) s_red[tid] += s_red[tid + off];
        __syncthreads();
    }
    float inv = 1.0f / s_red[0];
    __syncthreads();
    if (tid < HW) s_w[tid] = ex * inv;
    __syncthreads();

    // weighted context (coalesced over e)
    for (int e = tid; e < EMB; e += 256) {
        float s = 0.f;
        for (int hw = 0; hw < HW; ++hw)
            s += s_w[hw] * f[((size_t)b * HW + hw) * EMB + e];
        ctx[(size_t)b * EMB + e] = s;
    }
}

// ---------------------------------------------------------------------------
// LSTM cell pointwise (gate order i,f,g,o); b_hh folded in here.
// ---------------------------------------------------------------------------
__global__ void lstm_kernel(const float* __restrict__ gates,
                            const float* __restrict__ b_hh,
                            const float* __restrict__ cx_in,
                            float* __restrict__ hx_out,
                            float* __restrict__ cx_out)
{
    int idx = blockIdx.x * blockDim.x + threadIdx.x;
    if (idx >= BB * HID) return;
    int b = idx >> 10, j = idx & (HID - 1);
    const float* g = gates + (size_t)b * 4 * HID;
    float ig = g[j]           + b_hh[j];
    float fg = g[HID + j]     + b_hh[HID + j];
    float gg = g[2 * HID + j] + b_hh[2 * HID + j];
    float og = g[3 * HID + j] + b_hh[3 * HID + j];
    float si = 1.f / (1.f + __expf(-ig));
    float sf = 1.f / (1.f + __expf(-fg));
    float so = 1.f / (1.f + __expf(-og));
    float c  = sf * cx_in[idx] + si * tanhf(gg);
    cx_out[idx] = c;
    hx_out[idx] = so * tanhf(c);
}

// ---------------------------------------------------------------------------
// Host: enqueue full decoder dataflow on `stream` (graph-capture safe)
// ---------------------------------------------------------------------------
extern "C" void kernel_launch(void* const* d_in, const int* in_sizes, int n_in,
                              void* d_out, int out_size, void* d_ws, size_t ws_size,
                              hipStream_t stream)
{
    (void)in_sizes; (void)n_in; (void)out_size; (void)ws_size;

    const float* features = (const float*)d_in[0];
    const int*   captions = (const int*)  d_in[1];
    /* d_in[2] = lengths: uniform (== T), unused */
    const float* E      = (const float*)d_in[3];
    const float* W_feat = (const float*)d_in[4];
    const float* b_feat = (const float*)d_in[5];
    const float* W1     = (const float*)d_in[6];
    const float* b1     = (const float*)d_in[7];
    const float* W2     = (const float*)d_in[8];
    const float* b2     = (const float*)d_in[9];
    const float* Vv     = (const float*)d_in[10];
    const float* bV     = (const float*)d_in[11];
    const float* W_hi   = (const float*)d_in[12];
    const float* b_hi   = (const float*)d_in[13];
    const float* W_ci   = (const float*)d_in[14];
    const float* b_ci   = (const float*)d_in[15];
    const float* W_ih   = (const float*)d_in[16];
    const float* b_ih   = (const float*)d_in[17];
    const float* W_hh   = (const float*)d_in[18];
    const float* b_hh   = (const float*)d_in[19];
    const float* W_out  = (const float*)d_in[20];
    const float* b_out  = (const float*)d_in[21];

    float* ws = (float*)d_ws;
    size_t off = 0;
    auto alloc = [&](size_t nElem) { float* p = ws + off; off += nElem; return p; };
    float* f_buf   = alloc((size_t)BB * HW * EMB);   // 12544 x 512
    float* fa_buf  = alloc((size_t)BB * HW * ATT);   // 12544 x 512
    float* fmean   = alloc((size_t)BB * EMB);
    float* emb     = alloc((size_t)STEPS * BB * EMB);
    float* hxA     = alloc((size_t)BB * HID);
    float* hxB     = alloc((size_t)BB * HID);
    float* cxA     = alloc((size_t)BB * HID);
    float* cxB     = alloc((size_t)BB * HID);
    float* ha_buf  = alloc((size_t)BB * ATT);
    float* ctx_buf = alloc((size_t)BB * EMB);
    float* gates   = alloc((size_t)BB * 4 * HID);

    auto gemm = [&](const float* A, int lda, const float* Wm, int ldb,
                    const float* bias, float* C, int ldc,
                    int M, int N, int K, int beta) {
        dim3 grid((N + TN - 1) / TN, (M + TM - 1) / TM);
        gemm_wmma_bf16<<<grid, dim3(256), 0, stream>>>(
            A, lda, Wm, ldb, bias, C, ldc, M, N, K, beta);
    };

    // --- precompute -------------------------------------------------------
    embed_kernel<<<dim3(STEPS, BB), 128, 0, stream>>>(E, captions, emb);

    // f = features . W_feat^T + b_feat        [12544, 512]
    gemm(features, FEAT, W_feat, FEAT, b_feat, f_buf, EMB, BB * HW, EMB, FEAT, 0);
    mean_kernel<<<BB, 256, 0, stream>>>(f_buf, fmean);
    // fa = f . W1^T + b1                      [12544, 512]
    gemm(f_buf, EMB, W1, EMB, b1, fa_buf, ATT, BB * HW, ATT, EMB, 0);
    // hx0 / cx0                               [64, 1024]
    gemm(fmean, EMB, W_hi, EMB, b_hi, hxA, HID, BB, HID, EMB, 0);
    gemm(fmean, EMB, W_ci, EMB, b_ci, cxA, HID, BB, HID, EMB, 0);

    // --- recurrence -------------------------------------------------------
    float* hx_c = hxA; float* hx_n = hxB;
    float* cx_c = cxA; float* cx_n = cxB;
    for (int s = 0; s < STEPS; ++s) {
        // ha = hx . W2^T + b2                 [64, 512]
        gemm(hx_c, HID, W2, HID, b2, ha_buf, ATT, BB, ATT, HID, 0);
        // attention -> ctx                    [64, 512]
        attn_kernel<<<BB, 256, 0, stream>>>(fa_buf, ha_buf, f_buf, Vv, bV, ctx_buf);
        // gates = x.Wih[:, :512]^T + b_ih  +  ctx.Wih[:, 512:]^T  +  hx.Whh^T
        gemm(emb + (size_t)s * BB * EMB, EMB, W_ih, 2 * EMB, b_ih,
             gates, 4 * HID, BB, 4 * HID, EMB, 0);
        gemm(ctx_buf, EMB, W_ih + EMB, 2 * EMB, nullptr,
             gates, 4 * HID, BB, 4 * HID, EMB, 1);
        gemm(hx_c, HID, W_hh, HID, nullptr,
             gates, 4 * HID, BB, 4 * HID, HID, 1);
        // LSTM pointwise (adds b_hh)
        lstm_kernel<<<(BB * HID + 255) / 256, 256, 0, stream>>>(
            gates, b_hh, cx_c, hx_n, cx_n);
        // logits -> time-major output slice   [64, 10000]
        gemm(hx_n, HID, W_out, HID, b_out,
             (float*)d_out + (size_t)s * BB * VOCAB, VOCAB, BB, VOCAB, HID, 0);
        std::swap(hx_c, hx_n);
        std::swap(cx_c, cx_n);
    }
}
